// GCN_40647570489594
// MI455X (gfx1250) — matrix operand order repaired and checked by reading, not compiled, not run on hardware
//
#include <hip/hip_runtime.h>
#include <hip/hip_bf16.h>
#include <cstdint>

// ---------------------------------------------------------------------------
// GCN (3-layer, GCNConv + BN + ReLU + log_softmax) for MI455X / gfx1250.
// GEMMs use v_wmma_f32_16x16x32_bf16 (wave32, fp32 accumulate).
// Aggregation = self-loop analytic term + f32 atomic scatter (L2-resident).
// ---------------------------------------------------------------------------

typedef __attribute__((ext_vector_type(16))) __bf16 v16bf;
typedef __attribute__((ext_vector_type(8)))  float  v8f;

#define TPB 256

__device__ __forceinline__ unsigned short f2bf(float x) {
  // round-to-nearest-even float32 -> bf16 (storage only, no __bf16 arithmetic)
  unsigned u = __float_as_uint(x);
  u += 0x7FFFu + ((u >> 16) & 1u);
  return (unsigned short)(u >> 16);
}

// ---------------- utility kernels ----------------

__global__ void k_fill_f32(float* __restrict__ p, float v, long long n) {
  long long i = (long long)blockIdx.x * TPB + threadIdx.x;
  if (i < n) p[i] = v;
}

__global__ void k_deg_scatter(const long long* __restrict__ dst,
                              float* __restrict__ deg, int E) {
  int e = blockIdx.x * TPB + threadIdx.x;
  if (e < E) atomicAdd(&deg[(int)dst[e]], 1.0f);
}

__global__ void k_dinv(const float* __restrict__ deg, float* __restrict__ dinv, int n) {
  int i = blockIdx.x * TPB + threadIdx.x;
  if (i < n) dinv[i] = rsqrtf(fmaxf(deg[i], 1.0f));
}

// W [K x Fout] (row-major) -> Wt bf16 [colsPad x K], zero-padded columns
__global__ void k_wT_bf16(const float* __restrict__ W, unsigned short* __restrict__ Wt,
                          int K, int Fout, int colsPad) {
  int idx = blockIdx.x * TPB + threadIdx.x;
  int total = colsPad * K;
  if (idx >= total) return;
  int col = idx / K;
  int k   = idx - col * K;
  float v = (col < Fout) ? W[(size_t)k * Fout + col] : 0.0f;
  Wt[(size_t)col * K + k] = f2bf(v);
}

__global__ void k_f32_to_bf16(const float* __restrict__ a, unsigned short* __restrict__ o,
                              long long n) {
  long long i = (long long)blockIdx.x * TPB + threadIdx.x;
  if (i < n) o[i] = f2bf(a[i]);
}

// ---------------- WMMA bf16 GEMM: C[n x ncols] = A[n x K] * Bt^T ----------------
// A: bf16 row-major [nrows x K]; Bt: bf16 [colsPad x K] (i.e. W transposed)
// Block = 8 waves; wave tile = 16x16; block tile = 32 rows x 64 cols.
// Fragment layouts per CDNA5 ISA 7.12.2 (16-bit A 16x32, B 32x16, f32 C 16x16).
__global__ void k_gemm_bf16_wmma(const unsigned short* __restrict__ A,
                                 const unsigned short* __restrict__ Bt,
                                 float* __restrict__ C,
                                 int nrows, int K, int ldc, int ncols, int colsPad) {
  const int wave  = threadIdx.x >> 5;
  const int lane  = threadIdx.x & 31;
  const int laneM = lane & 15;     // M (for A) / N (for B,C) index within tile
  const int half  = lane >> 4;     // K-half selector
  const int rowBase = blockIdx.x * 32 + (wave & 1) * 16;
  const int colBase = blockIdx.y * 64 + (wave >> 1) * 16;
  if (colBase >= colsPad || rowBase >= nrows) return;   // uniform per wave

  const unsigned short* arow = A  + (size_t)(rowBase + laneM) * K;
  const unsigned short* bcol = Bt + (size_t)(colBase + laneM) * K;

  union { v16bf v; uint4 q[2]; } fa, fb;
  union { v8f  v; float f[8]; } acc;
#pragma unroll
  for (int r = 0; r < 8; ++r) acc.f[r] = 0.0f;

  for (int k0 = 0; k0 < K; k0 += 32) {
    // A lane holds row M=laneM: elems 0..7 -> K=k0+half*8+i, elems 8..15 -> K=k0+16+half*8+i
    fa.q[0] = *reinterpret_cast<const uint4*>(arow + k0 + half * 8);
    fa.q[1] = *reinterpret_cast<const uint4*>(arow + k0 + 16 + half * 8);
    // B lane holds col N=laneM: elems 0..7 -> K=k0+half*8+i, elems 8..15 -> K=k0+16+half*8+i
    fb.q[0] = *reinterpret_cast<const uint4*>(bcol + k0 + half * 8);
    fb.q[1] = *reinterpret_cast<const uint4*>(bcol + k0 + 16 + half * 8);
    acc.v = __builtin_amdgcn_wmma_f32_16x16x32_bf16(
        /*neg_a=*/false, fa.v, /*neg_b=*/false, fb.v,
        /*c_mod=*/(short)0, acc.v, /*reuse_a=*/false, /*reuse_b=*/false);
  }

  const int col = colBase + laneM;
  if (col < ncols) {
#pragma unroll
    for (int r = 0; r < 8; ++r) {              // C: VGPR r -> row M = r + 8*half
      int row = rowBase + r + half * 8;
      if (row < nrows) C[(size_t)row * ldc + col] = acc.f[r];
    }
  }
}

// ---------------- aggregation: out = D^-1/2 (A+I) D^-1/2 @ h + b ----------------

// self-loop + bias init: out[n,f] = h[n,f]*dinv[n]^2 + b[f]
__global__ void k_selfloop_bias(const float* __restrict__ h, const float* __restrict__ dinv,
                                const float* __restrict__ b, float* __restrict__ out,
                                long long total, int F) {
  long long i = (long long)blockIdx.x * TPB + threadIdx.x;
  if (i >= total) return;
  int f    = (int)(i % F);
  int node = (int)(i / F);
  float di = dinv[node];
  out[i] = h[i] * di * di + b[f];
}

// one wave per edge; atomicAdd into destination row (L2-resident table)
__global__ void k_edge_scatter(const long long* __restrict__ src,
                               const long long* __restrict__ dst,
                               const float* __restrict__ dinv,
                               const float* __restrict__ h,
                               float* __restrict__ out, int E, int F) {
  int e = blockIdx.x * 8 + (threadIdx.x >> 5);
  if (e >= E) return;
  int lane = threadIdx.x & 31;
  int s = (int)src[e], d = (int)dst[e];
  float w = dinv[s] * dinv[d];
  const float* hs = h   + (size_t)s * F;
  float*       od = out + (size_t)d * F;
  for (int f = lane; f < F; f += 32)
    atomicAdd(&od[f], hs[f] * w);
}

// ---------------- batchnorm (F == 256) ----------------

__global__ void k_bn_stats(const float* __restrict__ a, float* __restrict__ sum,
                           float* __restrict__ sumsq, int nrows) {
  int t  = threadIdx.x;                 // column 0..255
  int r0 = blockIdx.x * 256;
  int r1 = min(r0 + 256, nrows);
  float s = 0.0f, q = 0.0f;
  for (int r = r0; r < r1; ++r) {
    float v = a[(size_t)r * 256 + t];   // coalesced across threads
    s += v; q += v * v;
  }
  atomicAdd(&sum[t], s);
  atomicAdd(&sumsq[t], q);
}

__global__ void k_bn_finalize(const float* __restrict__ sum, const float* __restrict__ sumsq,
                              const float* __restrict__ g, const float* __restrict__ be,
                              float* __restrict__ scale, float* __restrict__ shift, int nrows) {
  int f = threadIdx.x;
  float inv_n = 1.0f / (float)nrows;
  float m = sum[f] * inv_n;
  float v = sumsq[f] * inv_n - m * m;
  float s = rsqrtf(v + 1e-5f) * g[f];
  scale[f] = s;
  shift[f] = be[f] - m * s;
}

// fused BN + ReLU + fp32->bf16 cast (F == 256)
__global__ void k_bn_relu_bf16(const float* __restrict__ a, const float* __restrict__ scale,
                               const float* __restrict__ shift, unsigned short* __restrict__ o,
                               long long n) {
  long long i = (long long)blockIdx.x * TPB + threadIdx.x;
  if (i >= n) return;
  int f = (int)(i & 255);
  float v = fmaf(a[i], scale[f], shift[f]);
  o[i] = f2bf(fmaxf(v, 0.0f));
}

// ---------------- log_softmax over C=40 columns, one wave per row ----------------

__global__ void k_log_softmax(float* __restrict__ out, int nrows, int C) {
  int row = blockIdx.x * 8 + (threadIdx.x >> 5);
  if (row >= nrows) return;
  int lane = threadIdx.x & 31;
  float* p = out + (size_t)row * C;
  float v0 = (lane      < C) ? p[lane]      : -INFINITY;
  float v1 = (lane + 32 < C) ? p[lane + 32] : -INFINITY;
  float m = fmaxf(v0, v1);
#pragma unroll
  for (int off = 16; off; off >>= 1) m = fmaxf(m, __shfl_xor(m, off));
  float s = ((lane      < C) ? expf(v0 - m) : 0.0f)
          + ((lane + 32 < C) ? expf(v1 - m) : 0.0f);
#pragma unroll
  for (int off = 16; off; off >>= 1) s += __shfl_xor(s, off);
  float lse = m + logf(s);
  if (lane      < C) p[lane]      = v0 - lse;
  if (lane + 32 < C) p[lane + 32] = v1 - lse;
}

// ---------------------------------------------------------------------------

static inline int cdiv(long long a, long long b) { return (int)((a + b - 1) / b); }

extern "C" void kernel_launch(void* const* d_in, const int* in_sizes, int n_in,
                              void* d_out, int out_size, void* d_ws, size_t ws_size,
                              hipStream_t stream) {
  const int N = in_sizes[0] / 128;       // 100000
  const int E = in_sizes[1] / 2;         // 3200000

  const float*     x    = (const float*)d_in[0];
  const long long* ei   = (const long long*)d_in[1];
  const long long* esrc = ei;
  const long long* edst = ei + E;
  const float* W1 = (const float*)d_in[2];  const float* b1  = (const float*)d_in[3];
  const float* W2 = (const float*)d_in[4];  const float* b2  = (const float*)d_in[5];
  const float* W3 = (const float*)d_in[6];  const float* b3  = (const float*)d_in[7];
  const float* g1 = (const float*)d_in[8];  const float* be1 = (const float*)d_in[9];
  const float* g2 = (const float*)d_in[10]; const float* be2 = (const float*)d_in[11];
  float* out = (float*)d_out;

  // workspace carve-up (256B aligned)
  char* ws = (char*)d_ws;
  size_t off = 0;
  auto take = [&](size_t bytes) -> char* {
    char* p = ws + off;
    off += (bytes + 255) & ~(size_t)255;
    return p;
  };
  float*          deg  = (float*)take((size_t)N * 4);
  float*          dinv = (float*)take((size_t)N * 4);
  unsigned short* w1t  = (unsigned short*)take((size_t)256 * 128 * 2);
  unsigned short* w2t  = (unsigned short*)take((size_t)256 * 256 * 2);
  unsigned short* w3t  = (unsigned short*)take((size_t)48  * 256 * 2);  // 40 cols padded to 48
  unsigned short* xb   = (unsigned short*)take((size_t)N * 128 * 2);
  float*          h    = (float*)take((size_t)N * 256 * 4);   // GEMM output (reused)
  float*          ag   = (float*)take((size_t)N * 256 * 4);   // aggregated (reused)
  unsigned short* hb   = (unsigned short*)take((size_t)N * 256 * 2);
  float*          bns  = (float*)take((size_t)512 * 4);       // [sum(256) | sumsq(256)]
  float*          bnq  = bns + 256;
  float*          bsc  = (float*)take((size_t)256 * 4);
  float*          bsh  = (float*)take((size_t)256 * 4);

  // ---- normalization factors: deg includes self-loop (init 1.0) ----
  k_fill_f32<<<cdiv(N, TPB), TPB, 0, stream>>>(deg, 1.0f, N);
  k_deg_scatter<<<cdiv(E, TPB), TPB, 0, stream>>>(edst, deg, E);
  k_dinv<<<cdiv(N, TPB), TPB, 0, stream>>>(deg, dinv, N);

  // ---- weight / feature conversion to bf16 ----
  k_wT_bf16<<<cdiv(256 * 128, TPB), TPB, 0, stream>>>(W1, w1t, 128, 256, 256);
  k_wT_bf16<<<cdiv(256 * 256, TPB), TPB, 0, stream>>>(W2, w2t, 256, 256, 256);
  k_wT_bf16<<<cdiv(48 * 256, TPB), TPB, 0, stream>>>(W3, w3t, 256, 40, 48);
  k_f32_to_bf16<<<cdiv((long long)N * 128, TPB), TPB, 0, stream>>>(x, xb, (long long)N * 128);

  const dim3 gemmGrid256(cdiv(N, 32), 4);   // 256 output cols
  const dim3 gemmGrid48(cdiv(N, 32), 1);    // 48 padded cols (layer 3)

  // ================= Layer 1: x@W1 -> agg -> BN -> ReLU =================
  k_gemm_bf16_wmma<<<gemmGrid256, TPB, 0, stream>>>(xb, w1t, h, N, 128, 256, 256, 256);
  k_selfloop_bias<<<cdiv((long long)N * 256, TPB), TPB, 0, stream>>>(h, dinv, b1, ag, (long long)N * 256, 256);
  k_edge_scatter<<<cdiv(E, 8), TPB, 0, stream>>>(esrc, edst, dinv, h, ag, E, 256);
  k_fill_f32<<<cdiv(512, TPB), TPB, 0, stream>>>(bns, 0.0f, 512);
  k_bn_stats<<<cdiv(N, 256), 256, 0, stream>>>(ag, bns, bnq, N);
  k_bn_finalize<<<1, 256, 0, stream>>>(bns, bnq, g1, be1, bsc, bsh, N);
  k_bn_relu_bf16<<<cdiv((long long)N * 256, TPB), TPB, 0, stream>>>(ag, bsc, bsh, hb, (long long)N * 256);

  // ================= Layer 2: h@W2 -> agg -> BN -> ReLU =================
  k_gemm_bf16_wmma<<<gemmGrid256, TPB, 0, stream>>>(hb, w2t, h, N, 256, 256, 256, 256);
  k_selfloop_bias<<<cdiv((long long)N * 256, TPB), TPB, 0, stream>>>(h, dinv, b2, ag, (long long)N * 256, 256);
  k_edge_scatter<<<cdiv(E, 8), TPB, 0, stream>>>(esrc, edst, dinv, h, ag, E, 256);
  k_fill_f32<<<cdiv(512, TPB), TPB, 0, stream>>>(bns, 0.0f, 512);
  k_bn_stats<<<cdiv(N, 256), 256, 0, stream>>>(ag, bns, bnq, N);
  k_bn_finalize<<<1, 256, 0, stream>>>(bns, bnq, g2, be2, bsc, bsh, N);
  k_bn_relu_bf16<<<cdiv((long long)N * 256, TPB), TPB, 0, stream>>>(ag, bsc, bsh, hb, (long long)N * 256);

  // ================= Layer 3: h@W3 -> agg -> log_softmax =================
  k_gemm_bf16_wmma<<<gemmGrid48, TPB, 0, stream>>>(hb, w3t, h, N, 256, 40, 40, 48);
  k_selfloop_bias<<<cdiv((long long)N * 40, TPB), TPB, 0, stream>>>(h, dinv, b3, out, (long long)N * 40, 40);
  k_edge_scatter<<<cdiv(E, 8), TPB, 0, stream>>>(esrc, edst, dinv, h, out, E, 40);
  k_log_softmax<<<cdiv(N, 8), TPB, 0, stream>>>(out, N, 40);
}
